// XAttention_32658931319258
// MI455X (gfx1250) — matrix operand order repaired and checked by reading, not compiled, not run on hardware
//
#include <hip/hip_runtime.h>

// ---------------------------------------------------------------------------
// Flash attention, N=2, L=S=4096, H=8, D=32, fp32 in/out, gfx1250 wave32.
// f16 WMMA (f32 accum). Scores computed transposed (K x Q^T) so exp'd probs
// land directly in the 16-bit A-operand layout for P x V (zero shuffles).
// exp2-domain softmax (log2e folded into Q scale). Double-buffered LDS with
// register prefetch: 1 barrier / 32-key chunk. 2 query tiles per wave so each
// LDS operand fetch feeds 2 WMMAs (8 WMMAs per chunk).
// ---------------------------------------------------------------------------

typedef __attribute__((ext_vector_type(16))) _Float16 v16h;
typedef __attribute__((ext_vector_type(8)))  _Float16 v8h;
typedef __attribute__((ext_vector_type(8)))  float    v8f;
typedef __attribute__((ext_vector_type(4)))  float    f32x4;

constexpr int N_ = 2, L_ = 4096, S_ = 4096, H_ = 8, D_ = 32;
constexpr int WAVES   = 4;                   // wave32s per block
constexpr int QTPW    = 2;                   // 16-row query tiles per wave
constexpr int BLOCK_Q = WAVES * QTPW * 16;   // 128 queries per block
constexpr int CHUNK   = 32;                  // keys per S-iteration
constexpr int KSTR    = 40;                  // LDS row stride in halves
constexpr int ITERS   = S_ / CHUNK;          // 128

__global__ __launch_bounds__(WAVES * 32)
void xattn_fa_kernel(const float* __restrict__ Q, const float* __restrict__ K,
                     const float* __restrict__ V, float* __restrict__ O)
{
    __shared__ _Float16 Klds[2][CHUNK * KSTR];   // [buf][key][dim]
    __shared__ _Float16 Vlds[2][D_    * KSTR];   // [buf][dim][key] (transposed)

    const int tid  = threadIdx.x;
    const int wave = tid >> 5;
    const int lane = tid & 31;
    const int lh   = lane & 15;
    const int hi   = lane >> 4;                  // 0 or 1

    const int nblocksL = L_ / BLOCK_Q;           // 32
    int bid = blockIdx.x;
    const int h    = bid % H_;
    int t2         = bid / H_;
    const int lblk = t2 % nblocksL;
    const int n    = t2 / nblocksL;
    const int l0   = lblk * BLOCK_Q + wave * (QTPW * 16);

    // 1/sqrt(32) * log2(e): scores live in the exp2 domain.
    const float qscale = 0.17677669529663687f * 1.4426950408889634f;

    // ---- Q as B-operand (32d x 16q): lane j -> query j&15, dims d0+e ----
    v16h bq[QTPW];
#pragma unroll
    for (int qt = 0; qt < QTPW; ++qt) {
        const int    qrow = l0 + qt * 16 + lh;
        const int    d0   = hi ? 16 : 0;
        const float* qp   = Q + ((size_t)(n * L_ + qrow) * H_ + h) * D_ + d0;
#pragma unroll
        for (int e = 0; e < 16; ++e) bq[qt][e] = (_Float16)(qp[e] * qscale);
    }

    v8f   o[QTPW][2];
    float m_run[QTPW], l_run[QTPW];
#pragma unroll
    for (int qt = 0; qt < QTPW; ++qt) {
        o[qt][0] = (v8f){}; o[qt][1] = (v8f){};
        m_run[qt] = -1e30f; l_run[qt] = 0.0f;
    }

    const size_t rowStride = (size_t)H_ * D_;                 // 256 floats
    const size_t kvBase    = (size_t)n * S_ * rowStride + (size_t)h * D_;
    const int    srow      = tid >> 2;                        // staging key row
    const int    c0        = (tid & 3) << 3;                  // staging dim col
    const size_t gOff      = kvBase + (size_t)srow * rowStride + c0;

    // ---- prefetch + stage chunk 0 into buffer 0 ----
    f32x4 pk0, pk1, pv0, pv1;
    {
        const float* kp = K + gOff;
        const float* vp = V + gOff;
        pk0 = *(const f32x4*)kp;  pk1 = *(const f32x4*)(kp + 4);
        pv0 = *(const f32x4*)vp;  pv1 = *(const f32x4*)(vp + 4);
        v8h kh;
#pragma unroll
        for (int i = 0; i < 4; ++i) { kh[i] = (_Float16)pk0[i]; kh[4 + i] = (_Float16)pk1[i]; }
        *(v8h*)(&Klds[0][srow * KSTR + c0]) = kh;
#pragma unroll
        for (int i = 0; i < 4; ++i) {
            Vlds[0][(c0 + i)     * KSTR + srow] = (_Float16)pv0[i];
            Vlds[0][(c0 + 4 + i) * KSTR + srow] = (_Float16)pv1[i];
        }
    }
    __syncthreads();

    for (int it = 0; it < ITERS; ++it) {
        const int  cur  = it & 1;
        const bool more = (it + 1) < ITERS;

        // ---- register prefetch of next chunk (hidden behind compute) ----
        if (more) {
            const size_t g  = gOff + (size_t)(it + 1) * CHUNK * rowStride;
            const float* kp = K + g;
            const float* vp = V + g;
            pk0 = *(const f32x4*)kp;  pk1 = *(const f32x4*)(kp + 4);
            pv0 = *(const f32x4*)vp;  pv1 = *(const f32x4*)(vp + 4);
        }

        const _Float16* Kb = Klds[cur];
        const _Float16* Vb = Vlds[cur];

        // ---- scores S^T = K x Q^T: one A fetch feeds both query tiles ----
        v8f st[QTPW][2];
#pragma unroll
        for (int sub = 0; sub < 2; ++sub) {
            const _Float16* kr = &Kb[(sub * 16 + lh) * KSTR + hi * 16];
            v16h ak;
#pragma unroll
            for (int e = 0; e < 8; ++e) { ak[e] = kr[e]; ak[8 + e] = kr[8 + e]; }
            v8f z = {};
#pragma unroll
            for (int qt = 0; qt < QTPW; ++qt)
                st[qt][sub] = __builtin_amdgcn_wmma_f32_16x16x32_f16(
                    false, ak, false, bq[qt], (short)0, z, false, false);
        }

        // ---- V B-operands (32k x 16d), loaded once, feed both tiles ----
        const _Float16* vr0 = &Vb[lh        * KSTR + hi * 8];
        const _Float16* vr1 = &Vb[(16 + lh) * KSTR + hi * 8];
        v16h bv0, bv1;
#pragma unroll
        for (int e = 0; e < 8; ++e) { bv0[e] = vr0[e]; bv0[8 + e] = vr0[16 + e]; bv1[e] = vr1[e]; bv1[8 + e] = vr1[16 + e]; }

#pragma unroll
        for (int qt = 0; qt < QTPW; ++qt) {
            // all 16 scores in a lane belong to query (lane&15) of this tile
            float mc = st[qt][0][0];
#pragma unroll
            for (int e = 0; e < 8; ++e) {
                mc = fmaxf(mc, st[qt][0][e]);
                mc = fmaxf(mc, st[qt][1][e]);
            }
            mc = fmaxf(mc, __shfl_xor(mc, 16, 32));
            const float m_new = fmaxf(m_run[qt], mc);
            const float alpha = __builtin_amdgcn_exp2f(m_run[qt] - m_new);

            // exp2 + pack straight into the P A-operand layout (no shuffles)
            v16h  ap;
            float psum = 0.0f;
#pragma unroll
            for (int e = 0; e < 8; ++e) {
                float p0 = __builtin_amdgcn_exp2f(st[qt][0][e] - m_new);
                float p1 = __builtin_amdgcn_exp2f(st[qt][1][e] - m_new);
                psum += p0 + p1;
                ap[e]     = (_Float16)p0;
                ap[8 + e] = (_Float16)p1;
            }
            psum += __shfl_xor(psum, 16, 32);
            l_run[qt] = l_run[qt] * alpha + psum;
            m_run[qt] = m_new;

            // rescale: row r in this lane is query r + 8*hi of the tile
            v8f o0s, o1s;
#pragma unroll
            for (int r = 0; r < 8; ++r) {
                float a_r = __shfl(alpha, hi * 8 + r, 32);
                o0s[r] = o[qt][0][r] * a_r;
                o1s[r] = o[qt][1][r] * a_r;
            }
            o[qt][0] = __builtin_amdgcn_wmma_f32_16x16x32_f16(
                false, ap, false, bv0, (short)0, o0s, false, false);
            o[qt][1] = __builtin_amdgcn_wmma_f32_16x16x32_f16(
                false, ap, false, bv1, (short)0, o1s, false, false);
        }

        // ---- stage prefetched chunk into the other buffer ----
        if (more) {
            _Float16* Kn = Klds[cur ^ 1];
            _Float16* Vn = Vlds[cur ^ 1];
            v8h kh;
#pragma unroll
            for (int i = 0; i < 4; ++i) { kh[i] = (_Float16)pk0[i]; kh[4 + i] = (_Float16)pk1[i]; }
            *(v8h*)(&Kn[srow * KSTR + c0]) = kh;
#pragma unroll
            for (int i = 0; i < 4; ++i) {
                Vn[(c0 + i)     * KSTR + srow] = (_Float16)pv0[i];
                Vn[(c0 + 4 + i) * KSTR + srow] = (_Float16)pv1[i];
            }
        }
        __syncthreads();
    }

    // ---- normalize + store ----
#pragma unroll
    for (int qt = 0; qt < QTPW; ++qt) {
#pragma unroll
        for (int r = 0; r < 8; ++r) {
            const int   qr   = r + hi * 8;
            const float linv = 1.0f / __shfl(l_run[qt], qr, 32);
            float* op = O + ((size_t)(n * L_ + (l0 + qt * 16 + qr)) * H_ + h) * D_;
            op[lh]      = o[qt][0][r] * linv;
            op[16 + lh] = o[qt][1][r] * linv;
        }
    }
}

extern "C" void kernel_launch(void* const* d_in, const int* in_sizes, int n_in,
                              void* d_out, int out_size, void* d_ws, size_t ws_size,
                              hipStream_t stream) {
    const float* Q = (const float*)d_in[0];
    const float* K = (const float*)d_in[1];
    const float* V = (const float*)d_in[2];
    float*       O = (float*)d_out;

    dim3 grid(N_ * H_ * (L_ / BLOCK_Q));   // 2*8*32 = 512 blocks
    dim3 block(WAVES * 32);                // 128 threads = 4 wave32s
    xattn_fa_kernel<<<grid, block, 0, stream>>>(Q, K, V, O);
}
